// FullAttention_13245679141297
// MI455X (gfx1250) — matrix operand order repaired and checked by reading, not compiled, not run on hardware
//
#include <hip/hip_runtime.h>
#include <math.h>

// ---------------------------------------------------------------------------
// Problem constants (from reference): x (B,T,C), H heads, d = C/H.
// ---------------------------------------------------------------------------
constexpr int B = 4;
constexpr int T = 2048;
constexpr int C = 1024;
constexpr int H = 16;
constexpr int D = 64;            // head dim
constexpr int M_ROWS = B * T;    // 8192 rows for the projection GEMMs
constexpr float SCALE = 0.125f;  // 1/sqrt(64)
constexpr int S_LD = 2052;       // padded LDS stride (floats): row offsets hit
                                 // distinct banks (2052 % 64 == 4)

typedef __bf16 bf16;
typedef __attribute__((ext_vector_type(16))) __bf16 v16bf;
typedef __attribute__((ext_vector_type(8)))  float  v8f;

// ---------------------------------------------------------------------------
// WMMA helper: D = A(16x32 bf16) x B(32x16 bf16) + C(16x16 f32)
// ---------------------------------------------------------------------------
__device__ __forceinline__ v8f wmma_bf16(v16bf a, v16bf b, v8f c) {
  return __builtin_amdgcn_wmma_f32_16x16x32_bf16(
      /*neg_a=*/false, a, /*neg_b=*/false, b,
      /*c_mod=*/(short)0, c, /*reuse_a=*/false, /*reuse_b=*/false);
}

// A-fragment (16x32, row-major source with leading dim ld, bf16):
// lanes 0-15 : row = lane,    K = k0+{0..7} and k0+{16..23}
// lanes 16-31: row = lane-16, K = k0+{8..15} and k0+{24..31}
__device__ __forceinline__ v16bf load_a_frag(const bf16* base, int ld, int k0, int lane) {
  const int half = lane >> 4, row = lane & 15;
  const bf16* p = base + (size_t)row * ld + k0 + half * 8;
  v16bf a;
#pragma unroll
  for (int j = 0; j < 8; ++j) { a[j] = p[j]; a[j + 8] = p[j + 16]; }
  return a;
}

// B-fragment (32x16): element (k,n) read from base[n*ld + k] (K contiguous
// per column n). lane n = lane&15; lanes 0-15 K=k0+{0..7,16..23}, lanes 16-31 K=k0+{8..15,24..31}.
__device__ __forceinline__ v16bf load_b_frag(const bf16* base, int ld, int k0, int lane) {
  const int half = lane >> 4, n = lane & 15;
  const bf16* p = base + (size_t)n * ld + k0 + half * 8;
  v16bf b;
#pragma unroll
  for (int j = 0; j < 8; ++j) { b[j] = p[j]; b[j + 8] = p[j + 16]; }
  return b;
}

// ---------------------------------------------------------------------------
// Kernel 0: fp32 -> bf16 conversion (grid-stride)
// ---------------------------------------------------------------------------
__global__ void cvt_f32_to_bf16(const float* __restrict__ src,
                                bf16* __restrict__ dst, int n) {
  int i = blockIdx.x * blockDim.x + threadIdx.x;
  int stride = gridDim.x * blockDim.x;
  for (; i < n; i += stride) dst[i] = (bf16)src[i];
}

// ---------------------------------------------------------------------------
// Shared GEMM core: 16x64 tile of  A(row-major, ld=C) @ Wmat^T, ping-pong
// double-buffered fragments (no register rotation copies).
// ---------------------------------------------------------------------------
__device__ __forceinline__ void gemm_16x64_core(
    const bf16* __restrict__ Arow, const bf16* __restrict__ W,
    int n0, int lane, v8f acc[4]) {
  v16bf a0 = load_a_frag(Arow, C, 0, lane);
  v16bf b0[4], b1[4];
#pragma unroll
  for (int nt = 0; nt < 4; ++nt)
    b0[nt] = load_b_frag(W + (size_t)(n0 + nt * 16) * C, C, 0, lane);

  for (int k0 = 0; k0 < C; k0 += 64) {           // 16 double-iterations
    // buffer 1 fragments for k0+32 in flight while buffer 0 computes
    v16bf a1 = load_a_frag(Arow, C, k0 + 32, lane);
#pragma unroll
    for (int nt = 0; nt < 4; ++nt)
      b1[nt] = load_b_frag(W + (size_t)(n0 + nt * 16) * C, C, k0 + 32, lane);
#pragma unroll
    for (int nt = 0; nt < 4; ++nt) acc[nt] = wmma_bf16(a0, b0[nt], acc[nt]);

    if (k0 + 64 < C) {                           // refill buffer 0 (in place)
      a0 = load_a_frag(Arow, C, k0 + 64, lane);
#pragma unroll
      for (int nt = 0; nt < 4; ++nt)
        b0[nt] = load_b_frag(W + (size_t)(n0 + nt * 16) * C, C, k0 + 64, lane);
    }
#pragma unroll
    for (int nt = 0; nt < 4; ++nt) acc[nt] = wmma_bf16(a1, b1[nt], acc[nt]);
  }
}

// ---------------------------------------------------------------------------
// Kernel 1: fused Q/K/V projection.  y = x @ W.T + b.
// One wave computes a 16x64 output tile.  grid = (M/16, C/64, 3).
// Q,K stored (B,H,T,d) bf16; V stored transposed (B,H,d,T) bf16 so the
// attention AV B-fragments are contiguous.
// ---------------------------------------------------------------------------
__global__ __launch_bounds__(32) void qkv_gemm(
    const bf16* __restrict__ xb, const bf16* __restrict__ wb,
    const float* __restrict__ bq, const float* __restrict__ bk,
    const float* __restrict__ bv,
    bf16* __restrict__ qout, bf16* __restrict__ kout, bf16* __restrict__ vout) {
  const int lane = threadIdx.x;
  const int m0 = blockIdx.x * 16;
  const int n0 = blockIdx.y * 64;
  const int mat = blockIdx.z;
  const bf16* W = wb + (size_t)mat * C * C;           // row n holds W[n, :]
  const float* bias = (mat == 0) ? bq : (mat == 1) ? bk : bv;

  v8f acc[4] = {v8f{}, v8f{}, v8f{}, v8f{}};
  gemm_16x64_core(xb + (size_t)m0 * C, W, n0, lane, acc);

  const int half = lane >> 4, nn = lane & 15;
#pragma unroll
  for (int nt = 0; nt < 4; ++nt) {
    const int ncol = n0 + nt * 16 + nn;
    const float bsv = bias[ncol];
    const int h = ncol >> 6, dch = ncol & 63;
#pragma unroll
    for (int r = 0; r < 8; ++r) {
      const int m = m0 + half * 8 + r;
      const int bb = m >> 11, t = m & 2047;   // m = b*T + t
      const float v = acc[nt][r] + bsv;
      if (mat == 0)
        qout[(((size_t)bb * H + h) * T + t) * D + dch] = (bf16)v;
      else if (mat == 1)
        kout[(((size_t)bb * H + h) * T + t) * D + dch] = (bf16)v;
      else
        vout[(((size_t)bb * H + h) * D + dch) * T + t] = (bf16)v;  // transposed
    }
  }
}

// ---------------------------------------------------------------------------
// Kernel 2: attention.  One block = 4 waves = one (b,h) 16-row query tile.
// Shared 16 x 2048 score strip (padded stride S_LD) + per-wave AV partials.
// grid = (B*H, T/16), block = 128.
//   QK^T : key tiles strided across the 4 waves (ping-pong fragment loads)
//   softmax: 8 lanes per row, strided columns (bank-conflict free)
//   att  : all 128 threads stream the tile to d_out coalesced
//   AV   : key axis split across waves, partials reduced through LDS
// ---------------------------------------------------------------------------
__global__ __launch_bounds__(128) void attn_kernel(
    const bf16* __restrict__ qb, const bf16* __restrict__ kb,
    const bf16* __restrict__ vtb, const float* __restrict__ mask,
    float* __restrict__ att_out, bf16* __restrict__ ybf) {
  extern __shared__ float smem[];
  float* s      = smem;                    // [16][S_LD] score strip
  float* invsum = smem + 16 * S_LD;        // [16]
  float* ypart  = invsum + 16;             // [4][16*64] AV partials

  const int tid  = threadIdx.x;
  const int w    = tid >> 5;               // wave id 0..3
  const int lane = tid & 31;
  const int half = lane >> 4, nn = lane & 15;
  const int bh = blockIdx.x;               // b*H + h
  const int bb = bh >> 4;
  const int h  = bh & 15;
  const int q0 = blockIdx.y * 16;

  const bf16* Q  = qb  + ((size_t)bh * T + q0) * D;  // 16 x 64 row-major
  const bf16* Kt = kb  + (size_t)bh * T * D;         // (key, d) row-major
  const bf16* Vt = vtb + (size_t)bh * D * T;         // (d, key) row-major

  // Q A-fragments for K-dim chunks 0-31 and 32-63 (registers, per wave).
  const v16bf qa0 = load_a_frag(Q, D, 0, lane);
  const v16bf qa1 = load_a_frag(Q, D, 32, lane);

  // ---- S = Q K^T * scale : wave w handles key tiles kt = w, w+4, ... -----
  // 32 tiles per wave, processed in ping-pong pairs.
  {
    auto kblk = [&](int kt) { return Kt + (size_t)kt * 16 * D; };
    auto emit = [&](int kt, v16bf bA, v16bf bB) {
      v8f c = v8f{};
      c = wmma_bf16(qa0, bA, c);
      c = wmma_bf16(qa1, bB, c);
#pragma unroll
      for (int r = 0; r < 8; ++r)
        s[(half * 8 + r) * S_LD + kt * 16 + nn] = c[r] * SCALE;
    };
    v16bf b0a = load_b_frag(kblk(w), D, 0, lane);
    v16bf b0b = load_b_frag(kblk(w), D, 32, lane);
    for (int i = 0; i < 32; i += 2) {
      const int ta = w + 4 * i;            // tile for buffer 0
      const int tb = ta + 4;               // tile for buffer 1 (i+1 <= 31)
      v16bf b1a = load_b_frag(kblk(tb), D, 0, lane);
      v16bf b1b = load_b_frag(kblk(tb), D, 32, lane);
      emit(ta, b0a, b0b);
      if (i + 2 < 32) {                    // refill buffer 0 in place
        b0a = load_b_frag(kblk(ta + 8), D, 0, lane);
        b0b = load_b_frag(kblk(ta + 8), D, 32, lane);
      }
      emit(tb, b1a, b1b);
    }
  }
  __syncthreads();

  // ---- softmax over keys: row = 4*w + (lane&3), 8 lanes per row ----------
  {
    const int row = 4 * w + (lane & 3);
    const int seg = lane >> 2;             // 0..7
    float* sr = s + (size_t)row * S_LD;
    float m = -1e30f;
    for (int i = 0; i < T / 8; ++i) m = fmaxf(m, sr[seg + 8 * i]);
    m = fmaxf(m, __shfl_xor(m, 4));
    m = fmaxf(m, __shfl_xor(m, 8));
    m = fmaxf(m, __shfl_xor(m, 16));
    float sum = 0.0f;
    for (int i = 0; i < T / 8; ++i) {
      const int col = seg + 8 * i;
      const float e = __expf(sr[col] - m);
      sr[col] = e;
      sum += e;
    }
    sum += __shfl_xor(sum, 4);
    sum += __shfl_xor(sum, 8);
    sum += __shfl_xor(sum, 16);
    if (seg == 0) invsum[row] = 1.0f / sum;
  }
  __syncthreads();

  // ---- normalize, apply post-softmax key mask, stream att to d_out -------
  const float* mrow = mask + (size_t)bb * T;
  float* attp = att_out + ((size_t)bh * T + q0) * T;
  for (int idx = tid; idx < 16 * T; idx += 128) {    // coalesced over keys
    const int row = idx >> 11, col = idx & (T - 1);
    const float p = s[(size_t)row * S_LD + col] * invsum[row] * mrow[col];
    s[(size_t)row * S_LD + col] = p;                 // reused as AV A-matrix
    attp[(size_t)row * T + col] = p;
  }
  __syncthreads();

  // ---- Y = P @ V : wave w reduces keys [w*512, w*512+512) ----------------
  v8f y[4] = {v8f{}, v8f{}, v8f{}, v8f{}};
  {
    const int kbeg = w * (T / 4), kend = kbeg + T / 4;  // 16 chunks of 32
    auto load_pa = [&](int kk) {           // A-frag from LDS fp32 -> bf16
      v16bf a;
      const float* pr = s + (size_t)nn * S_LD + kk + half * 8;
#pragma unroll
      for (int j = 0; j < 8; ++j) { a[j] = (bf16)pr[j]; a[j + 8] = (bf16)pr[j + 16]; }
      return a;
    };
    v16bf a0 = load_pa(kbeg);
    v16bf b0[4], b1[4];
#pragma unroll
    for (int nt = 0; nt < 4; ++nt)
      b0[nt] = load_b_frag(Vt + (size_t)(nt * 16) * T, T, kbeg, lane);

    for (int kk = kbeg; kk < kend; kk += 64) {       // 8 double-iterations
      v16bf a1 = load_pa(kk + 32);
#pragma unroll
      for (int nt = 0; nt < 4; ++nt)
        b1[nt] = load_b_frag(Vt + (size_t)(nt * 16) * T, T, kk + 32, lane);
#pragma unroll
      for (int nt = 0; nt < 4; ++nt) y[nt] = wmma_bf16(a0, b0[nt], y[nt]);

      if (kk + 64 < kend) {                          // refill buffer 0
        a0 = load_pa(kk + 64);
#pragma unroll
        for (int nt = 0; nt < 4; ++nt)
          b0[nt] = load_b_frag(Vt + (size_t)(nt * 16) * T, T, kk + 64, lane);
      }
#pragma unroll
      for (int nt = 0; nt < 4; ++nt) y[nt] = wmma_bf16(a1, b1[nt], y[nt]);
    }
  }

  // store per-wave partials, then cooperative cross-wave reduction
#pragma unroll
  for (int nt = 0; nt < 4; ++nt)
#pragma unroll
    for (int r = 0; r < 8; ++r)
      ypart[w * (16 * 64) + (half * 8 + r) * 64 + nt * 16 + nn] = y[nt][r];
  __syncthreads();

  for (int idx = tid; idx < 16 * 64; idx += 128) {
    const float acc = ypart[idx] + ypart[1024 + idx] +
                      ypart[2048 + idx] + ypart[3072 + idx];
    const int r = idx >> 6, cc = idx & 63;
    const int t = q0 + r;
    ybf[((size_t)bb * T + t) * C + h * D + cc] = (bf16)acc;
  }
}

// ---------------------------------------------------------------------------
// Kernel 3: output projection y_out = ybf @ Wp^T + bp  (fp32 out to d_out)
// ---------------------------------------------------------------------------
__global__ __launch_bounds__(32) void proj_gemm(
    const bf16* __restrict__ ab, const bf16* __restrict__ Wp,
    const float* __restrict__ bp, float* __restrict__ out) {
  const int lane = threadIdx.x;
  const int m0 = blockIdx.x * 16;
  const int n0 = blockIdx.y * 64;

  v8f acc[4] = {v8f{}, v8f{}, v8f{}, v8f{}};
  gemm_16x64_core(ab + (size_t)m0 * C, Wp, n0, lane, acc);

  const int half = lane >> 4, nn = lane & 15;
#pragma unroll
  for (int nt = 0; nt < 4; ++nt) {
    const int ncol = n0 + nt * 16 + nn;
    const float bsv = bp[ncol];
#pragma unroll
    for (int r = 0; r < 8; ++r) {
      const int m = m0 + half * 8 + r;
      out[(size_t)m * C + ncol] = acc[nt][r] + bsv;
    }
  }
}

// ---------------------------------------------------------------------------
// Host-side launcher
// ---------------------------------------------------------------------------
extern "C" void kernel_launch(void* const* d_in, const int* in_sizes, int n_in,
                              void* d_out, int out_size, void* d_ws, size_t ws_size,
                              hipStream_t stream) {
  const float* x    = (const float*)d_in[0];
  // d_in[1] = encoder_output (unused by reference forward)
  const float* mask = (const float*)d_in[2];
  const float* Wq = (const float*)d_in[3];  const float* bq = (const float*)d_in[4];
  const float* Wk = (const float*)d_in[5];  const float* bk = (const float*)d_in[6];
  const float* Wv = (const float*)d_in[7];  const float* bv = (const float*)d_in[8];
  const float* Wp = (const float*)d_in[9];  const float* bp = (const float*)d_in[10];

  float* y_out   = (float*)d_out;                         // (B,T,C) fp32
  float* att_out = y_out + (size_t)B * T * C;             // (B,H,T,T) fp32

  // Workspace layout (bf16 elements): 44M elems = 88 MB.
  const size_t BTC = (size_t)B * T * C;                   // 8M
  bf16* xb  = (bf16*)d_ws;                                // x in bf16
  bf16* wb  = xb + BTC;                                   // 4 weight mats (q,k,v,p)
  bf16* qb  = wb + 4ull * C * C;                          // (B,H,T,d)
  bf16* kb  = qb + BTC;                                   // (B,H,T,d)
  bf16* vtb = kb + BTC;                                   // (B,H,d,T) transposed
  bf16* ybf = vtb + BTC;                                  // (B,T,C) attn output

  // 0) fp32 -> bf16 conversions (x + 4 weight matrices)
  cvt_f32_to_bf16<<<1024, 256, 0, stream>>>(x, xb, (int)BTC);
  cvt_f32_to_bf16<<<256, 256, 0, stream>>>(Wq, wb + 0ull * C * C, C * C);
  cvt_f32_to_bf16<<<256, 256, 0, stream>>>(Wk, wb + 1ull * C * C, C * C);
  cvt_f32_to_bf16<<<256, 256, 0, stream>>>(Wv, wb + 2ull * C * C, C * C);
  cvt_f32_to_bf16<<<256, 256, 0, stream>>>(Wp, wb + 3ull * C * C, C * C);

  // 1) Q/K/V projections: grid (M/16, C/64, 3), one wave per 16x64 tile.
  qkv_gemm<<<dim3(M_ROWS / 16, C / 64, 3), 32, 0, stream>>>(
      xb, wb, bq, bk, bv, qb, kb, vtb);

  // 2) Attention: grid (B*H, T/16), 4 waves/block, ~144 KB dynamic LDS.
  const size_t smem_bytes = (16ull * S_LD + 16 + 4 * 16 * 64) * sizeof(float);
  attn_kernel<<<dim3(B * H, T / 16), 128, smem_bytes, stream>>>(
      qb, kb, vtb, mask, att_out, ybf);

  // 3) Output projection to fp32 d_out.
  proj_gemm<<<dim3(M_ROWS / 16, C / 64), 32, 0, stream>>>(
      ybf, wb + 3ull * C * C, bp, y_out);
}
